// EnhancedFractionalPINO_48661979464032
// MI455X (gfx1250) — matrix operand order repaired and checked by reading, not compiled, not run on hardware
//
#include <hip/hip_runtime.h>
#include <math.h>

// ---------------------------------------------------------------------------
// EnhancedFractionalPINO forward for MI455X (gfx1250, wave32, WMMA, TDM).
//
//   K1  fft2 real part per 64x64 image (LDS DFT)              -> fr_pre
//   K2  pack z = s + i*w (w_j via lgamma closed form)         -> bufA
//   K3  Stockham FFT: 1x radix-2 + 11x radix-4 passes (fwd)   -> Z
//   K4  two-for-one spectral separation, S*W*scale            -> bufP
//   K3  1x radix-2 + 11x radix-4 passes (inverse, 1/L folded) -> conv
//   K5  real part -> bf16 activation
//   K6  weights f32 -> bf16
//   K7  5x GEMM: A-tile staged via TENSOR_LOAD_TO_LDS (TDM) + s_wait_tensorcnt,
//       B-tile transposed in LDS, v_wmma_f32_16x16x32_bf16, f32 accumulate
//   K8  ifft2 real part per image -> d_out
// ---------------------------------------------------------------------------

#define BATCH 256
#define CH    3
#define MODES 12288                 // C*H*W
#define WIDTH 512
#define NSIG  (BATCH * MODES)       // 3,145,728
#define LOG2L 23
#define LFFT  (1 << LOG2L)          // 8,388,608 >= 2*NSIG-1 -> linear conv
#define TFFT  (LFFT / 2)
#define QFFT  (LFFT / 4)
#define PI_F  3.14159265358979323846f

typedef __bf16 bf16_t;
typedef __attribute__((ext_vector_type(16))) __bf16 v16bf;
typedef __attribute__((ext_vector_type(8)))  float  v8f;
typedef __attribute__((ext_vector_type(4)))  unsigned int u32x4_t;
typedef __attribute__((ext_vector_type(8)))  int i32x8_t;
typedef __attribute__((ext_vector_type(4)))  int i32x4_t;

#if defined(__has_builtin)
#  if __has_builtin(__builtin_amdgcn_tensor_load_to_lds)
#    define HAVE_TDM 1
#  else
#    define HAVE_TDM 0
#  endif
#else
#  define HAVE_TDM 0
#endif

static __device__ __forceinline__ bf16_t f2bf(float f) {
  union { float f; unsigned int u; } v; v.f = f;
  unsigned int r = v.u + 0x7FFFu + ((v.u >> 16) & 1u);   // round-to-nearest-even
  union { unsigned short u; bf16_t b; } o;
  o.u = (unsigned short)(r >> 16);
  return o.b;
}

#if HAVE_TDM
// Issue one TDM descriptor: load a 64-row x 32-col bf16 tile (row stride = K
// elements) from global into LDS at lds_byte_addr, rows packed contiguously.
static __device__ __forceinline__ void tdm_load_a_tile(
    const bf16_t* gptr, unsigned int lds_byte_addr, int K) {
  const unsigned long long ga = (unsigned long long)(size_t)gptr;
  u32x4_t g0;
  g0[0] = 1u;                                            // count=1 (user D#)
  g0[1] = lds_byte_addr;                                 // lds_addr
  g0[2] = (unsigned int)(ga & 0xFFFFFFFFu);              // global_addr[31:0]
  g0[3] = (unsigned int)((ga >> 32) & 0x01FFFFFFu) | (2u << 30); // addr|type=2
  i32x8_t g1;
  g1[0] = (int)(1u << 16);                               // data_size=1 (2B)
  g1[1] = (int)(((unsigned int)K & 0xFFFFu) << 16);      // tensor_dim0 lo16
  g1[2] = (int)(((unsigned int)K >> 16) | (256u << 16)); // dim0 hi | tensor_dim1=256
  g1[3] = (int)(32u << 16);                              // tile_dim0 = 32
  g1[4] = 64;                                            // tile_dim1 = 64
  g1[5] = K;                                             // tensor_dim0_stride
  g1[6] = 0;
  g1[7] = 0;
  i32x4_t z4 = {0, 0, 0, 0};
#if __has_include(<hip/amd_detail/amd_gfx1250_TDM.h>)
  i32x8_t z8 = {0, 0, 0, 0, 0, 0, 0, 0};
  __builtin_amdgcn_tensor_load_to_lds(g0, g1, z4, z4, z8, 0);
#else
  __builtin_amdgcn_tensor_load_to_lds(g0, g1, z4, z4, 0);
#endif
}
#endif

// ------------------------- K1: fft2, keep real part -------------------------
__global__ __launch_bounds__(256) void fft2_real_kernel(
    const float* __restrict__ x, float* __restrict__ fr) {
  const int img = blockIdx.x;                  // 0 .. B*C-1
  const float* xi = x + (size_t)img * 4096;
  __shared__ float xs[4096];
  __shared__ float yr[4096];
  __shared__ float yi[4096];
  __shared__ float twc[64], tws[64];
  const int tid = threadIdx.x;
  if (tid < 64) {
    float a = (2.0f * PI_F / 64.0f) * (float)tid;
    __sincosf(a, &tws[tid], &twc[tid]);
  }
  for (int i = tid; i < 4096; i += 256) xs[i] = xi[i];
  __syncthreads();
  // Y[u][w] = sum_h x[h][w] * e^{-2*pi*i*u*h/64}
  for (int i = tid; i < 4096; i += 256) {
    const int u = i >> 6, w = i & 63;
    float sr = 0.0f, si = 0.0f;
    for (int h = 0; h < 64; ++h) {
      const int idx = (u * h) & 63;
      const float xv = xs[(h << 6) | w];
      sr += xv * twc[idx];
      si -= xv * tws[idx];
    }
    yr[i] = sr; yi[i] = si;
  }
  __syncthreads();
  // Re F[u][v] = sum_w yr*cos + yi*sin (angle 2pi v w/64)
  for (int i = tid; i < 4096; i += 256) {
    const int u = i >> 6, v = i & 63;
    float s = 0.0f;
    for (int w = 0; w < 64; ++w) {
      const int idx = (v * w) & 63;
      s += yr[(u << 6) | w] * twc[idx] + yi[(u << 6) | w] * tws[idx];
    }
    fr[(size_t)img * 4096 + i] = s;
  }
}

// -------- K2: pack z = s + i*w ; w_j = Gamma(j-a)/(Gamma(-a)Gamma(j+1)) -----
__global__ __launch_bounds__(256) void build_z_kernel(
    const float* __restrict__ fr, float2* __restrict__ z) {
  const int i = blockIdx.x * 256 + threadIdx.x;
  float s = 0.0f, w = 0.0f;
  if (i < NSIG) {
    s = fr[i];
    if (i == 0) w = 1.0f;
    else        w = -0.28209479177387814f *     // 1/|Gamma(-0.5)|
                    expf(lgammaf((float)i - 0.5f) - lgammaf((float)i + 1.0f));
  }
  z[i] = make_float2(s, w);
}

// ----------------- K3a: one radix-2 Stockham pass ---------------------------
__global__ __launch_bounds__(256) void fft_pass2_kernel(
    const float2* __restrict__ in, float2* __restrict__ out,
    int lgNs, float angStep) {
  const int t  = blockIdx.x * 256 + threadIdx.x;     // 0 .. L/2-1
  const int Ns = 1 << lgNs;
  const int j  = t & (Ns - 1);
  const int g  = t >> lgNs;
  const float2 a = in[t];
  const float2 b = in[t + TFFT];
  float sn, cs;
  __sincosf(angStep * (float)j, &sn, &cs);
  const float2 wb = make_float2(cs * b.x - sn * b.y, cs * b.y + sn * b.x);
  const int o = (g << (lgNs + 1)) + j;
  out[o]      = make_float2(a.x + wb.x, a.y + wb.y);
  out[o + Ns] = make_float2(a.x - wb.x, a.y - wb.y);
}

// ----------------- K3b: one radix-4 Stockham pass ---------------------------
// angStep = sign * 2*pi / (4*Ns) ; inv != 0 for the inverse transform.
__global__ __launch_bounds__(256) void fft_pass4_kernel(
    const float2* __restrict__ in, float2* __restrict__ out,
    int lgNs, float angStep, int inv) {
  const int t  = blockIdx.x * 256 + threadIdx.x;     // 0 .. L/4-1
  const int Ns = 1 << lgNs;
  const int j  = t & (Ns - 1);
  const int g  = t >> lgNs;
  const float2 a0 = in[t];
  const float2 a1 = in[t + QFFT];
  const float2 a2 = in[t + 2 * QFFT];
  const float2 a3 = in[t + 3 * QFFT];
  float s1, c1;
  __sincosf(angStep * (float)j, &s1, &c1);
  const float c2 = c1 * c1 - s1 * s1, s2 = 2.0f * c1 * s1;   // tw1^2
  const float c3 = c2 * c1 - s2 * s1, s3 = c2 * s1 + s2 * c1; // tw1^3
  const float2 b1 = make_float2(c1 * a1.x - s1 * a1.y, c1 * a1.y + s1 * a1.x);
  const float2 b2 = make_float2(c2 * a2.x - s2 * a2.y, c2 * a2.y + s2 * a2.x);
  const float2 b3 = make_float2(c3 * a3.x - s3 * a3.y, c3 * a3.y + s3 * a3.x);
  const float2 e0 = make_float2(a0.x + b2.x, a0.y + b2.y);
  const float2 e1 = make_float2(a0.x - b2.x, a0.y - b2.y);
  const float2 e2 = make_float2(b1.x + b3.x, b1.y + b3.y);
  const float2 e3 = make_float2(b1.x - b3.x, b1.y - b3.y);
  // forward: r3 = -i*e3 ; inverse: r3 = +i*e3
  const float2 r3 = inv ? make_float2(-e3.y, e3.x) : make_float2(e3.y, -e3.x);
  const int o = (g << (lgNs + 2)) + j;
  out[o]          = make_float2(e0.x + e2.x, e0.y + e2.y);
  out[o + Ns]     = make_float2(e1.x + r3.x, e1.y + r3.y);
  out[o + 2 * Ns] = make_float2(e0.x - e2.x, e0.y - e2.y);
  out[o + 3 * Ns] = make_float2(e1.x - r3.x, e1.y - r3.y);
}

// ------- K4: two-for-one separation S,W from Z; P = S*W*scale ---------------
__global__ __launch_bounds__(256) void spectral_kernel(
    const float2* __restrict__ Z, float2* __restrict__ P, float factor) {
  const int k  = blockIdx.x * 256 + threadIdx.x;
  const int kr = (LFFT - k) & (LFFT - 1);
  const float2 a  = Z[k];
  const float2 bc = Z[kr];
  const float2 b  = make_float2(bc.x, -bc.y);          // conj(Z[-k])
  const float2 S  = make_float2(0.5f * (a.x + b.x), 0.5f * (a.y + b.y));
  const float2 D  = make_float2(a.x - b.x, a.y - b.y);
  const float2 W  = make_float2(0.5f * D.y, -0.5f * D.x);  // (Z-conj)/2i
  const float2 p  = make_float2(S.x * W.x - S.y * W.y, S.x * W.y + S.y * W.x);
  P[k] = make_float2(p.x * factor, p.y * factor);
}

// ----------------------- K5: take real part -> bf16 -------------------------
__global__ __launch_bounds__(256) void extract_kernel(
    const float2* __restrict__ conv, bf16_t* __restrict__ outb) {
  const int i = blockIdx.x * 256 + threadIdx.x;
  if (i < NSIG) outb[i] = f2bf(conv[i].x);
}

// ----------------------- K6: f32 -> bf16 weight convert ---------------------
__global__ __launch_bounds__(256) void f32_to_bf16_kernel(
    const float* __restrict__ in, bf16_t* __restrict__ out, int n) {
  const int i = blockIdx.x * 256 + threadIdx.x;
  if (i < n) out[i] = f2bf(in[i]);
}

// ----------------------- K7: WMMA bf16 GEMM ---------------------------------
// C(256 x Nn) = A(256 x K) * W(K x Nn) + bias ; optional ReLU.
// WG = 256 thr (8 wave32) computes 64 x 128 tile; wave -> 16 x 64 strip.
// A-tile staged by TDM (tensor_load_to_lds), B-tile transposed via LDS stores.
union alignas(32) Frag8 {
  unsigned int u[8];
  uint4 q[2];
  v16bf v;
};

template<bool RELU, bool OUTF32>
__global__ __launch_bounds__(256) void gemm_bf16_kernel(
    const bf16_t* __restrict__ A, const bf16_t* __restrict__ W,
    const float* __restrict__ bias,
    bf16_t* __restrict__ outB, float* __restrict__ outF,
    int K, int Nn) {
  __shared__ alignas(16) bf16_t As[64][32];    // [m][k]
  __shared__ alignas(16) bf16_t Bs[128][32];   // [n][k] (transposed stage)
  const int tid  = threadIdx.x;
  const int wave = tid >> 5;
  const int lane = tid & 31;
  const int m_off  = (wave & 3) * 16;
  const int n_base = (wave >> 2) * 64;
  const int bm = blockIdx.x * 64;
  const int bn = blockIdx.y * 128;

  // stage-in assignments
  const int ar  = tid >> 2;            // 0..63   A row in tile (fallback path)
  const int ac  = (tid & 3) * 8;       // 0..24   A k-chunk (8 bf16 = 16B)
  const int bk  = (tid >> 4) * 2;      // 0..30   B k row pair
  const int bnc = (tid & 15) * 8;      // 0..120  B n-chunk (8 bf16 = 16B)
  (void)ar; (void)ac;

  const int hlf = lane >> 4;           // 0/1
  const int l15 = lane & 15;

  v8f acc[4];
#pragma unroll
  for (int s = 0; s < 4; ++s)
#pragma unroll
    for (int r = 0; r < 8; ++r) acc[s][r] = 0.0f;

  for (int k0 = 0; k0 < K; k0 += 32) {
    // coalesced global reads of B rows (registers only; LDS still in use)
    const bf16_t* wrow = W + (size_t)(k0 + bk) * Nn + (bn + bnc);
    const uint4 b0 = *(const uint4*)(wrow);
    const uint4 b1 = *(const uint4*)(wrow + Nn);
#if !HAVE_TDM
    const uint4 av = *(const uint4*)(A + (size_t)(bm + ar) * K + (k0 + ac));
#endif
    if (k0 + 32 < K)
      __builtin_prefetch(W + (size_t)(k0 + 32 + bk) * Nn + (bn + bnc), 0, 1);
    __syncthreads();                   // previous iteration done reading LDS
#if HAVE_TDM
    if (tid == 0)
      tdm_load_a_tile(A + (size_t)bm * K + k0,
                      (unsigned int)(size_t)&As[0][0], K);
#else
    *(uint4*)&As[ar][ac] = av;
#endif
    // transpose B into [n][k] with paired (k, k+1) dword stores
    const unsigned short* p0 = (const unsigned short*)&b0;
    const unsigned short* p1 = (const unsigned short*)&b1;
#pragma unroll
    for (int j = 0; j < 8; ++j) {
      const unsigned int pr = (unsigned int)p0[j] | ((unsigned int)p1[j] << 16);
      *(unsigned int*)&Bs[bnc + j][bk] = pr;
    }
#if HAVE_TDM
    if (tid == 0) __builtin_amdgcn_s_wait_tensorcnt(0);  // TDM tile landed
#endif
    __syncthreads();

    // A fragment (16x32 bf16): lane<16 row M=l, K in {0..7,16..23};
    // lane>=16 row M=l-16, K in {8..15,24..31}  -> two b128 LDS reads.
    Frag8 af;
    af.q[0] = *(const uint4*)&As[m_off + l15][hlf * 8];
    af.q[1] = *(const uint4*)&As[m_off + l15][16 + hlf * 8];
#pragma unroll
    for (int s = 0; s < 4; ++s) {
      // B fragment (32x16 bf16): lane<16 col N=l, K in {0..7,16..23}; lane>=16 K in {8..15,24..31}.
      Frag8 bfr;
      bfr.q[0] = *(const uint4*)&Bs[n_base + s * 16 + l15][hlf * 8];
      bfr.q[1] = *(const uint4*)&Bs[n_base + s * 16 + l15][16 + hlf * 8];
      acc[s] = __builtin_amdgcn_wmma_f32_16x16x32_bf16(
          false, af.v, false, bfr.v, (short)0, acc[s], false, false);
    }
  }

  // epilogue: C layout VGPR r, lane l -> M = r + 8*(l>=16), N = l%16
#pragma unroll
  for (int s = 0; s < 4; ++s) {
    const int n = bn + n_base + s * 16 + l15;
    const float bb = bias[n];
#pragma unroll
    for (int r = 0; r < 8; ++r) {
      const int m = bm + m_off + hlf * 8 + r;
      float vv = acc[s][r] + bb;
      if (RELU) vv = fmaxf(vv, 0.0f);
      if (OUTF32) outF[(size_t)m * Nn + n] = vv;
      else        outB[(size_t)m * Nn + n] = f2bf(vv);
    }
  }
}

// ----------------------- K8: ifft2, keep real part --------------------------
__global__ __launch_bounds__(256) void ifft2_real_kernel(
    const float* __restrict__ proc, float* __restrict__ out) {
  const int img = blockIdx.x;
  const float* p = proc + (size_t)img * 4096;
  __shared__ float ps[4096];
  __shared__ float yr[4096];
  __shared__ float yi[4096];
  __shared__ float twc[64], tws[64];
  const int tid = threadIdx.x;
  if (tid < 64) {
    float a = (2.0f * PI_F / 64.0f) * (float)tid;
    __sincosf(a, &tws[tid], &twc[tid]);
  }
  for (int i = tid; i < 4096; i += 256) ps[i] = p[i];
  __syncthreads();
  // Y[h][v] = sum_u p[u][v] * e^{+2 pi i u h/64}
  for (int i = tid; i < 4096; i += 256) {
    const int h = i >> 6, v = i & 63;
    float sr = 0.0f, si = 0.0f;
    for (int u = 0; u < 64; ++u) {
      const int idx = (u * h) & 63;
      const float pv = ps[(u << 6) | v];
      sr += pv * twc[idx];
      si += pv * tws[idx];
    }
    yr[(h << 6) | v] = sr;
    yi[(h << 6) | v] = si;
  }
  __syncthreads();
  // out[h][w] = (1/4096) * sum_v ( yr*cos - yi*sin )(2 pi v w/64)
  for (int i = tid; i < 4096; i += 256) {
    const int h = i >> 6, w = i & 63;
    float s = 0.0f;
    for (int v = 0; v < 64; ++v) {
      const int idx = (v * w) & 63;
      s += yr[(h << 6) | v] * twc[idx] - yi[(h << 6) | v] * tws[idx];
    }
    out[(size_t)img * 4096 + i] = s * (1.0f / 4096.0f);
  }
}

// ---------------------------------------------------------------------------
extern "C" void kernel_launch(void* const* d_in, const int* in_sizes, int n_in,
                              void* d_out, int out_size, void* d_ws, size_t ws_size,
                              hipStream_t stream) {
  (void)in_sizes; (void)out_size;
  if (n_in < 11) return;
  const float* x   = (const float*)d_in[0];
  const float* Ws1 = (const float*)d_in[1];
  const float* bs1 = (const float*)d_in[2];
  const float* Ws2 = (const float*)d_in[3];
  const float* bs2 = (const float*)d_in[4];
  const float* Wn1 = (const float*)d_in[5];
  const float* bn1 = (const float*)d_in[6];
  const float* Wn2 = (const float*)d_in[7];
  const float* bn2 = (const float*)d_in[8];
  const float* Wn3 = (const float*)d_in[9];
  const float* bn3 = (const float*)d_in[10];
  float* out = (float*)d_out;

  char* ws = (char*)d_ws;
  size_t off = 0;
  auto take = [&](size_t bytes) -> void* {
    void* p = ws + off;
    off += (bytes + 255) & ~(size_t)255;
    return p;
  };
  float*  fr_pre = (float*)  take((size_t)NSIG * 4);
  float2* bufA   = (float2*) take((size_t)LFFT * 8);
  float2* bufB   = (float2*) take((size_t)LFFT * 8);
  float2* bufP   = (float2*) take((size_t)LFFT * 8);
  bf16_t* Ws1b = (bf16_t*) take((size_t)MODES * WIDTH * 2);
  bf16_t* Ws2b = (bf16_t*) take((size_t)MODES * WIDTH * 2);
  bf16_t* Wn1b = (bf16_t*) take((size_t)MODES * WIDTH * 2);
  bf16_t* Wn2b = (bf16_t*) take((size_t)WIDTH * WIDTH * 2);
  bf16_t* Wn3b = (bf16_t*) take((size_t)MODES * WIDTH * 2);
  bf16_t* frA  = (bf16_t*) take((size_t)NSIG * 2);
  bf16_t* spec = (bf16_t*) take((size_t)NSIG * 2);
  bf16_t* t1   = (bf16_t*) take((size_t)BATCH * WIDTH * 2);
  bf16_t* t2   = (bf16_t*) take((size_t)BATCH * WIDTH * 2);
  bf16_t* t3   = (bf16_t*) take((size_t)BATCH * WIDTH * 2);
  float*  proc = (float*)  take((size_t)NSIG * 4);
  if (off > ws_size) return;   // deterministic guard

  // 1) fft2 real part
  fft2_real_kernel<<<BATCH * CH, 256, 0, stream>>>(x, fr_pre);

  // 2) pack z = s + i*w, zero-padded to L
  build_z_kernel<<<LFFT / 256, 256, 0, stream>>>(fr_pre, bufA);

  // 3) forward FFT: radix-2 (Ns=1), then 11x radix-4
  float2* ping = bufA; float2* pong = bufB;
  fft_pass2_kernel<<<TFFT / 256, 256, 0, stream>>>(ping, pong, 0, -PI_F);
  { float2* t = ping; ping = pong; pong = t; }
  for (int p = 0; p < 11; ++p) {
    const int lg = 1 + 2 * p;                       // Ns = 2 * 4^p
    fft_pass4_kernel<<<QFFT / 256, 256, 0, stream>>>(
        ping, pong, lg, -(PI_F * 0.5f) / (float)(1 << lg), 0);
    float2* t = ping; ping = pong; pong = t;
  }
  // Z now in `ping`; `pong` is free.

  // 4) spectral product (scale = h^{-alpha} / L)
  const float factor = sqrtf((float)(NSIG - 1)) / (float)LFFT;
  spectral_kernel<<<LFFT / 256, 256, 0, stream>>>(ping, bufP, factor);

  // 5) inverse FFT: radix-2, then 11x radix-4
  float2* ip = bufP; float2* op = pong;
  fft_pass2_kernel<<<TFFT / 256, 256, 0, stream>>>(ip, op, 0, PI_F);
  { float2* t = ip; ip = op; op = t; }
  for (int p = 0; p < 11; ++p) {
    const int lg = 1 + 2 * p;
    fft_pass4_kernel<<<QFFT / 256, 256, 0, stream>>>(
        ip, op, lg, (PI_F * 0.5f) / (float)(1 << lg), 1);
    float2* t = ip; ip = op; op = t;
  }

  // 6) real part -> bf16 activation
  extract_kernel<<<NSIG / 256, 256, 0, stream>>>(ip, frA);

  // 7) weights -> bf16
  f32_to_bf16_kernel<<<(MODES * WIDTH + 255) / 256, 256, 0, stream>>>(Ws1, Ws1b, MODES * WIDTH);
  f32_to_bf16_kernel<<<(MODES * WIDTH + 255) / 256, 256, 0, stream>>>(Ws2, Ws2b, MODES * WIDTH);
  f32_to_bf16_kernel<<<(MODES * WIDTH + 255) / 256, 256, 0, stream>>>(Wn1, Wn1b, MODES * WIDTH);
  f32_to_bf16_kernel<<<(WIDTH * WIDTH + 255) / 256, 256, 0, stream>>>(Wn2, Wn2b, WIDTH * WIDTH);
  f32_to_bf16_kernel<<<(MODES * WIDTH + 255) / 256, 256, 0, stream>>>(Wn3, Wn3b, MODES * WIDTH);

  // 8) GEMM chain
  gemm_bf16_kernel<true,  false><<<dim3(BATCH / 64, WIDTH / 128), 256, 0, stream>>>(
      frA, Ws1b, bs1, t1, nullptr, MODES, WIDTH);
  gemm_bf16_kernel<false, false><<<dim3(BATCH / 64, MODES / 128), 256, 0, stream>>>(
      t1, Ws2b, bs2, spec, nullptr, WIDTH, MODES);
  gemm_bf16_kernel<true,  false><<<dim3(BATCH / 64, WIDTH / 128), 256, 0, stream>>>(
      spec, Wn1b, bn1, t2, nullptr, MODES, WIDTH);
  gemm_bf16_kernel<true,  false><<<dim3(BATCH / 64, WIDTH / 128), 256, 0, stream>>>(
      t2, Wn2b, bn2, t3, nullptr, WIDTH, WIDTH);
  gemm_bf16_kernel<false, true ><<<dim3(BATCH / 64, MODES / 128), 256, 0, stream>>>(
      t3, Wn3b, bn3, nullptr, proc, WIDTH, MODES);

  // 9) ifft2 real part -> output
  ifft2_real_kernel<<<BATCH * CH, 256, 0, stream>>>(proc, out);
}
